// ChebConv_20203526160488
// MI455X (gfx1250) — compile-verified, matrix operand order checked by reading, not run on hardware
//
#include <hip/hip_runtime.h>
#include <hip/hip_bf16.h>
#include <stdint.h>

#define N_NODES 50000
#define N_EDGES 640000
#define WIDTH   128
#define KCHEB   5
#define N_CONVS 6

typedef float v2f  __attribute__((ext_vector_type(2)));
typedef float v8f  __attribute__((ext_vector_type(8)));
typedef _Float16 v16h __attribute__((ext_vector_type(16)));

// ---------------------------------------------------------------------------
// Elementwise helpers
// ---------------------------------------------------------------------------
__global__ void k_zero(float* __restrict__ p, int n) {
    int i = blockIdx.x * blockDim.x + threadIdx.x;
    if (i < n) p[i] = 0.0f;
}

// dest[i] = alpha * src[i]
__global__ void k_init_scaled(float* __restrict__ dest, const float* __restrict__ src,
                              float alpha, int n) {
    int i = blockIdx.x * blockDim.x + threadIdx.x;
    if (i < n) dest[i] = alpha * src[i];
}

// acc[node*128 + c] = bias[c]
__global__ void k_init_bias(float* __restrict__ acc, const float* __restrict__ bias, int n) {
    int i = blockIdx.x * blockDim.x + threadIdx.x;
    if (i < n) acc[i] = bias[i & (WIDTH - 1)];
}

__global__ void k_relu(float* __restrict__ dst, const float* __restrict__ src, int n) {
    int i = blockIdx.x * blockDim.x + threadIdx.x;
    if (i < n) dst[i] = fmaxf(src[i], 0.0f);
}

// ---------------------------------------------------------------------------
// Graph normalization
// ---------------------------------------------------------------------------
__global__ void k_degree(float* __restrict__ deg, const long long* __restrict__ dst, int nE) {
    int e = blockIdx.x * blockDim.x + threadIdx.x;
    if (e < nE) atomicAdd(&deg[dst[e]], 1.0f);
}

__global__ void k_dinv(float* __restrict__ dinv, const float* __restrict__ deg, int n) {
    int i = blockIdx.x * blockDim.x + threadIdx.x;
    if (i < n) {
        float d = deg[i];
        dinv[i] = (d > 0.0f) ? rsqrtf(fmaxf(d, 1e-12f)) : 0.0f;
    }
}

__global__ void k_norm(float* __restrict__ norm, const long long* __restrict__ src,
                       const long long* __restrict__ dst, const float* __restrict__ dinv, int nE) {
    int e = blockIdx.x * blockDim.x + threadIdx.x;
    if (e < nE) norm[e] = -dinv[src[e]] * dinv[dst[e]];
}

// ---------------------------------------------------------------------------
// Propagate: dest[dst] += scale * norm[e] * x[src]   (dest pre-initialized)
// One wave (32 lanes) per edge, float4 (4 channels) per lane.
// ---------------------------------------------------------------------------
__global__ __launch_bounds__(256) void k_scatter(float* __restrict__ dest,
                                                 const float* __restrict__ x,
                                                 const long long* __restrict__ src,
                                                 const long long* __restrict__ dst,
                                                 const float* __restrict__ norm,
                                                 float scale, int nE) {
    int t = blockIdx.x * blockDim.x + threadIdx.x;
    int e = t >> 5;
    if (e >= nE) return;
    int lane = t & 31;
    long long s = src[e];
    long long d = dst[e];
    float w = scale * norm[e];
    const float4* xp = (const float4*)(x + (size_t)s * WIDTH);
    float4 v = xp[lane];
    float* dp = dest + (size_t)d * WIDTH + lane * 4;
    atomicAdd(dp + 0, w * v.x);
    atomicAdd(dp + 1, w * v.y);
    atomicAdd(dp + 2, w * v.z);
    atomicAdd(dp + 3, w * v.w);
}

// ---------------------------------------------------------------------------
// GEMM-accumulate: acc[M x 128] += T[M x 128] @ W[128 x 128]
// Block = 256 threads = 8 waves; one 16-row strip per block staged in LDS;
// wave w owns N-tile [16w, 16w+16); 32 WMMA f32 16x16x4 steps over K=128.
// 50000 rows = 3125 * 16 exactly -> EXEC all-ones everywhere (WMMA requirement).
// ---------------------------------------------------------------------------
__global__ __launch_bounds__(256) void k_gemm_acc(float* __restrict__ acc,
                                                  const float* __restrict__ T,
                                                  const float* __restrict__ W) {
    __shared__ float ldsT[16 * WIDTH];
    const int m0  = blockIdx.x * 16;
    const int tid = threadIdx.x;

    // Coalesced stage of the 16x128 strip (8 floats per thread).
    for (int i = tid; i < 16 * WIDTH; i += 256) {
        ldsT[i] = T[(size_t)(m0 + (i >> 7)) * WIDTH + (i & 127)];
    }
    __syncthreads();

    const int wave = tid >> 5;      // 0..7 -> N tile
    const int lane = tid & 31;
    const int n0   = wave * 16;
    const int l    = lane & 15;
    const int hi   = lane >> 4;     // 0: lanes 0-15, 1: lanes 16-31

    // C/D layout: VGPR r -> M = r (lanes 0-15) or M = 8+r (lanes 16-31), N = l.
    v8f c;
    {
        const float* cp = acc + (size_t)(m0 + hi * 8) * WIDTH + n0 + l;
        #pragma unroll
        for (int r = 0; r < 8; ++r) c[r] = cp[(size_t)r * WIDTH];
    }

#if __has_builtin(__builtin_amdgcn_wmma_f32_16x16x4_f32)
    // fp32 WMMA: A 16x4 (2 VGPR), B 4x16 (2 VGPR).
    // A: lane l (0-15): K = kk+2*hi, kk+2*hi+1 for row M=l.  B mirrors with N=l.
    #pragma unroll
    for (int kk = 0; kk < WIDTH; kk += 4) {
        const int ka = kk + hi * 2;
        v2f a, b;
        a.x = ldsT[l * WIDTH + ka];
        a.y = ldsT[l * WIDTH + ka + 1];
        b.x = W[(size_t)ka * WIDTH + n0 + l];
        b.y = W[(size_t)(ka + 1) * WIDTH + n0 + l];
        c = __builtin_amdgcn_wmma_f32_16x16x4_f32(false, a, false, b,
                                                  (short)0, c, false, false);
    }
#else
    // Fallback: f16 inputs, fp32 accumulate (codegen-confirmed builtin).
    #pragma unroll
    for (int kb = 0; kb < WIDTH; kb += 32) {
        v16h a, b;
        #pragma unroll
        for (int j = 0; j < 16; ++j) {
            const int v = j >> 1, h = j & 1;
            const int k = (v < 4) ? (hi * 8 + v * 2 + h)
                                  : (16 + hi * 8 + (v - 4) * 2 + h);
            a[j] = (_Float16)ldsT[l * WIDTH + kb + k];
            b[j] = (_Float16)W[(size_t)(kb + k) * WIDTH + n0 + l];
        }
        c = __builtin_amdgcn_wmma_f32_16x16x32_f16(false, a, false, b,
                                                   (short)0, c, false, false);
    }
#endif

    float* cp = acc + (size_t)(m0 + hi * 8) * WIDTH + n0 + l;
    #pragma unroll
    for (int r = 0; r < 8; ++r) cp[(size_t)r * WIDTH] = c[r];
}

// ---------------------------------------------------------------------------
// Final Linear(128 -> 1): one wave per node, shuffle reduction.
// ---------------------------------------------------------------------------
__global__ __launch_bounds__(256) void k_final_lin(float* __restrict__ out,
                                                   const float* __restrict__ h,
                                                   const float* __restrict__ lw,
                                                   const float* __restrict__ lb, int nN) {
    int t = blockIdx.x * blockDim.x + threadIdx.x;
    int node = t >> 5;
    if (node >= nN) return;
    int lane = t & 31;
    const float4* hp = (const float4*)(h + (size_t)node * WIDTH);
    const float4* wp = (const float4*)lw;
    float4 hv = hp[lane];
    float4 wv = wp[lane];
    float s = hv.x * wv.x + hv.y * wv.y + hv.z * wv.z + hv.w * wv.w;
    #pragma unroll
    for (int o = 16; o > 0; o >>= 1) s += __shfl_xor(s, o, 32);
    if (lane == 0) out[node] = s + lb[0];
}

// ---------------------------------------------------------------------------
// Host orchestration
// ---------------------------------------------------------------------------
extern "C" void kernel_launch(void* const* d_in, const int* in_sizes, int n_in,
                              void* d_out, int out_size, void* d_ws, size_t ws_size,
                              hipStream_t stream) {
    const float*     x      = (const float*)d_in[0];
    const long long* ei     = (const long long*)d_in[1];   // [2, E] int64
    const float*     conv_w = (const float*)d_in[2];       // [6,5,128,128]
    const float*     conv_b = (const float*)d_in[3];       // [6,128]
    const float*     lin_w  = (const float*)d_in[4];       // [128]
    const float*     lin_b  = (const float*)d_in[5];       // [1]
    float*           out    = (float*)d_out;

    const long long* e_src = ei;
    const long long* e_dst = ei + N_EDGES;

    // Carve workspace (256-byte aligned chunks).
    char* ws = (char*)d_ws;
    auto carve = [&](size_t nbytes) {
        char* p = ws;
        ws += (nbytes + 255) & ~(size_t)255;
        return p;
    };
    const size_t NF = (size_t)N_NODES * WIDTH;
    float* H    = (float*)carve(NF * 4);
    float* ACC  = (float*)carve(NF * 4);
    float* Ta   = (float*)carve(NF * 4);
    float* Tb   = (float*)carve(NF * 4);
    float* deg  = (float*)carve((size_t)N_NODES * 4);
    float* dinv = (float*)carve((size_t)N_NODES * 4);
    float* norm = (float*)carve((size_t)N_EDGES * 4);
    (void)ws_size; (void)in_sizes; (void)n_in; (void)out_size;

    const int TB = 256;
    const int gNF   = (int)((NF + TB - 1) / TB);                 // element grids
    const int gN    = (N_NODES + TB - 1) / TB;
    const int gE    = (N_EDGES + TB - 1) / TB;
    const int gEdge = (int)(((size_t)N_EDGES * 32 + TB - 1) / TB); // wave-per-edge
    const int gGemm = N_NODES / 16;                               // 3125, exact
    const int gLin  = (int)(((size_t)N_NODES * 32 + TB - 1) / TB);

    // --- normalization: deg -> d^-1/2 -> per-edge weight ---
    k_zero<<<gN, TB, 0, stream>>>(deg, N_NODES);
    k_degree<<<gE, TB, 0, stream>>>(deg, e_dst, N_EDGES);
    k_dinv<<<gN, TB, 0, stream>>>(dinv, deg, N_NODES);
    k_norm<<<gE, TB, 0, stream>>>(norm, e_src, e_dst, dinv, N_EDGES);

    // --- ChebConv stack ---
    for (int l = 0; l < N_CONVS; ++l) {
        const float* in = (l == 0) ? x : H;
        const float* Wl = conv_w + (size_t)l * KCHEB * WIDTH * WIDTH;
        const float* bl = conv_b + (size_t)l * WIDTH;

        // out = bias + T0 @ W0           (T0 = in)
        k_init_bias<<<gNF, TB, 0, stream>>>(ACC, bl, (int)NF);
        k_gemm_acc<<<gGemm, TB, 0, stream>>>(ACC, in, Wl + 0 * WIDTH * WIDTH);

        // T1 = L @ in  -> Ta
        k_zero<<<gNF, TB, 0, stream>>>(Ta, (int)NF);
        k_scatter<<<gEdge, TB, 0, stream>>>(Ta, in, e_src, e_dst, norm, 1.0f, N_EDGES);
        k_gemm_acc<<<gGemm, TB, 0, stream>>>(ACC, Ta, Wl + 1 * WIDTH * WIDTH);

        // T2 = 2 L T1 - T0  -> Tb   (init Tb = -in, scatter 2*norm*Ta)
        k_init_scaled<<<gNF, TB, 0, stream>>>(Tb, in, -1.0f, (int)NF);
        k_scatter<<<gEdge, TB, 0, stream>>>(Tb, Ta, e_src, e_dst, norm, 2.0f, N_EDGES);
        k_gemm_acc<<<gGemm, TB, 0, stream>>>(ACC, Tb, Wl + 2 * WIDTH * WIDTH);

        // T3 = 2 L T2 - T1  -> H buffer (layer input dead by now)
        k_init_scaled<<<gNF, TB, 0, stream>>>(H, Ta, -1.0f, (int)NF);
        k_scatter<<<gEdge, TB, 0, stream>>>(H, Tb, e_src, e_dst, norm, 2.0f, N_EDGES);
        k_gemm_acc<<<gGemm, TB, 0, stream>>>(ACC, H, Wl + 3 * WIDTH * WIDTH);

        // T4 = 2 L T3 - T2  -> Ta (old T1 dead)
        k_init_scaled<<<gNF, TB, 0, stream>>>(Ta, Tb, -1.0f, (int)NF);
        k_scatter<<<gEdge, TB, 0, stream>>>(Ta, H, e_src, e_dst, norm, 2.0f, N_EDGES);
        k_gemm_acc<<<gGemm, TB, 0, stream>>>(ACC, Ta, Wl + 4 * WIDTH * WIDTH);

        if (l < N_CONVS - 1) {
            k_relu<<<gNF, TB, 0, stream>>>(H, ACC, (int)NF);
        }
    }

    // Final Linear(128, 1) on last conv output (no activation).
    k_final_lin<<<gLin, TB, 0, stream>>>(out, ACC, lin_w, lin_b, N_NODES);
}
